// ConvLSTM_70944269796171
// MI455X (gfx1250) — matrix operand order repaired and checked by reading, not compiled, not run on hardware
//
#include <hip/hip_runtime.h>
#include <hip/hip_bf16.h>
#include <cstdint>
#include <cstddef>

typedef __attribute__((ext_vector_type(16))) _Float16 v16h;
typedef __attribute__((ext_vector_type(8)))  _Float16 v8h;
typedef __attribute__((ext_vector_type(8)))  float    v8f;

#define DEVINL __device__ __forceinline__

DEVINL float hsig(float x) { return fminf(fmaxf(0.2f * x + 0.5f, 0.f), 1.f); }

DEVINL v8f wmma16x16x32(v16h a, v16h b, v8f c) {
  // emits v_wmma_f32_16x16x32_f16
  return __builtin_amdgcn_wmma_f32_16x16x32_f16(false, a, false, b, (short)0, c,
                                                false, false);
}

DEVINL v16h zero_frag() {
  v16h z;
#pragma unroll
  for (int j = 0; j < 16; ++j) z[j] = (_Float16)0.f;
  return z;
}

// A-fragment (16x32 f16) for v_wmma_f32_16x16x32_f16 built from an LDS
// row-major [128][C] f16 image, im2col contraction index K = tap*C + ch.
// Per ISA 7.12.2: lane (l&15) supplies row M; elems 0..7 -> K = (l>>4)*8+j,
// elems 8..15 -> K = 16 + (l>>4)*8 + (j-8).  C % 32 == 0 so a 32-wide K
// chunk never crosses a tap boundary -> two contiguous 16B LDS loads.
DEVINL v16h a_frag_lds(const _Float16* p, int C, int klbase, int arow, int half) {
  int tap = klbase / C;
  int chb = klbase - tap * C;
  int rm  = arow + tap - 1;          // SAME padding, kernel width 3
  if (rm < 0 || rm >= 128) return zero_frag();
  union { v16h v; v8h h2[2]; } u;
  u.h2[0] = *(const v8h*)(p + rm * C + chb + half * 8);
  u.h2[1] = *(const v8h*)(p + rm * C + chb + 16 + half * 8);
  return u.v;
}

// ---------------------------------------------------------------------------
// Weight pre-swizzle into B-fragment order: dst[kc][nt][lane][16] f16 where
// n = nt*16 + (lane&15), k = kc*32 + (lane>>4)*8 + j (j<8), kc*32 + 16 + (lane>>4)*8 + (j-8) (j>=8)  (ISA B 32x16 layout).
// RNN variant fuses [Wx (K padded to KXP) ; Wh] along K.
// ---------------------------------------------------------------------------
__global__ void prep_rnn_w(const float* __restrict__ Wx, const float* __restrict__ Wh,
                           _Float16* __restrict__ dst, int CI, int F, int KXP,
                           size_t total) {
  const int N4 = 4 * F;
  const int NT = N4 / 16;
  for (size_t idx = (size_t)blockIdx.x * blockDim.x + threadIdx.x; idx < total;
       idx += (size_t)gridDim.x * blockDim.x) {
    int e     = (int)(idx & 511);
    int lane  = e >> 4, j = e & 15;
    size_t fi = idx >> 9;
    int nt    = (int)(fi % NT);
    int kc    = (int)(fi / NT);
    int n     = nt * 16 + (lane & 15);
    int k     = kc * 32 + ((j < 8) ? ((lane >> 4) << 3) + j : 16 + ((lane >> 4) << 3) + (j - 8));
    float v = 0.f;
    if (k < KXP) {
      if (k < 3 * CI) v = Wx[(size_t)k * N4 + n];       // Wx [3,CI,4F] flat
    } else {
      int kh = k - KXP;                                  // < 3F
      v = Wh[(size_t)kh * N4 + n];                       // Wh [3,F,4F] flat
    }
    dst[idx] = (_Float16)v;
  }
}

__global__ void prep_dense_w(const float* __restrict__ W, _Float16* __restrict__ dst,
                             int K, int N, int NPAD, size_t total) {
  const int NT = NPAD / 16;
  for (size_t idx = (size_t)blockIdx.x * blockDim.x + threadIdx.x; idx < total;
       idx += (size_t)gridDim.x * blockDim.x) {
    int e     = (int)(idx & 511);
    int lane  = e >> 4, j = e & 15;
    size_t fi = idx >> 9;
    int nt    = (int)(fi % NT);
    int kc    = (int)(fi / NT);
    int n     = nt * 16 + (lane & 15);
    int k     = kc * 32 + ((j < 8) ? ((lane >> 4) << 3) + j : 16 + ((lane >> 4) << 3) + (j - 8));
    dst[idx] = (_Float16)((n < N) ? W[(size_t)k * N + n] : 0.f);
  }
}

// ---------------------------------------------------------------------------
// Fused ConvLSTM1D layer. One workgroup per batch element; 8 waves = 8
// M-tiles of L=128.  gates[L,4F] = [im2col(x_t)|im2col(h_{t-1})] @ Wcat + b
// via v_wmma_f32_16x16x32_f16.  Each wave's N-group holds the i/f/c/o tiles
// of the SAME 16 channels, so the LSTM pointwise update fuses in registers.
// h is double-buffered in LDS (ping-pong per timestep); c streams via L2.
// ---------------------------------------------------------------------------
template <int CI, int F, int KXP, bool LAST>
__global__ __launch_bounds__(256) void convlstm_kernel(
    const float* __restrict__ x0,        // [B,T,128] fp32 (CI==1)
    const _Float16* __restrict__ xseq,   // [B,T,128,CI] f16 (CI>1)
    const _Float16* __restrict__ Wf,     // swizzled [KC][4F/16][32][16] f16
    const float* __restrict__ bias,      // [4F], gate-major i,f,c,o
    float* __restrict__ cws,             // [B,128,F] f32 cell state
    _Float16* __restrict__ hseq,         // [B,T,128,F] f16 (!LAST)
    float* __restrict__ hlast,           // [B,128,F] f32 (LAST)
    int B, int T) {
  constexpr int L    = 128;
  constexpr int NF   = F / 16;           // channel tiles
  constexpr int NT   = 4 * NF;           // gate-major N tiles
  constexpr int KC_X = KXP / 32;
  constexpr int KC_H = (3 * F) / 32;
  constexpr int KC   = KC_X + KC_H;
  constexpr int XN   = (CI == 1) ? 132 : L * CI;

  __shared__ __align__(16) _Float16 sh[2][L * F];  // h ping-pong
  __shared__ __align__(16) _Float16 sx[XN];        // staged x_t

  const int b    = blockIdx.x;
  const int tid  = threadIdx.x;
  const int lane = tid & 31;
  const int mt   = tid >> 5;             // wave id == M tile
  const int half = lane >> 4;
  const int col  = lane & 15;
  const int arow = mt * 16 + col;        // A row this lane supplies

  const v16h* __restrict__ Wv = (const v16h*)Wf;

  for (int i = tid; i < L * F; i += 256) sh[0][i] = (_Float16)0.f;  // h(0)=0
  __syncthreads();

  for (int t = 0; t < T; ++t) {
    // ---- stage x_t into LDS (f16) ----
    if constexpr (CI == 1) {
      for (int i = tid; i < 130; i += 256) {
        int r = i - 1;                                   // halo x[-1]=x[L]=0
        sx[i] = (r >= 0 && r < L) ? (_Float16)x0[((size_t)b * T + t) * L + r]
                                  : (_Float16)0.f;
      }
    } else {
      const _Float16* xp = xseq + ((size_t)b * T + t) * L * CI;
      for (int i = tid * 8; i < L * CI; i += 256 * 8)
        *(v8h*)(sx + i) = *(const v8h*)(xp + i);
    }
    __syncthreads();

    const _Float16* hcur = sh[t & 1];
    _Float16* hnxt       = &sh[(t + 1) & 1][0];

    for (int ng = 0; ng < NF; ++ng) {
      const int f = ng * 16 + col;       // output channel for this lane
      v8f acc[4];
#pragma unroll
      for (int g = 0; g < 4; ++g) {
        float bv = bias[g * F + f];
#pragma unroll
        for (int r = 0; r < 8; ++r) acc[g][r] = bv;
      }

      for (int kc = 0; kc < KC; ++kc) {
        v16h af;
        if (kc < KC_X) {
          if constexpr (CI == 1) {       // K-block zero-padded 3 -> 32
            af = zero_frag();
#pragma unroll
            for (int j = 0; j < 16; ++j) {
              int ko = (j < 8) ? (half * 8 + j) : (16 + half * 8 + (j - 8));
              if (ko < 3) af[j] = sx[arow + ko];   // sx[r+1] = x[r], halo=0
            }
          } else {
            af = a_frag_lds(sx, CI, kc * 32, arow, half);
          }
        } else {
          af = a_frag_lds(hcur, F, (kc - KC_X) * 32, arow, half);
        }
        const size_t kb = (size_t)kc * NT * 32;
#pragma unroll
        for (int g = 0; g < 4; ++g) {    // i, f, c, o tiles share the A frag
          v16h bf = Wv[kb + (size_t)(g * NF + ng) * 32 + lane];
          acc[g]  = wmma16x16x32(af, bf, acc[g]);
        }
      }

      // ---- fused LSTM pointwise update (keras: hard_sigmoid gates, relu) ---
#pragma unroll
      for (int r = 0; r < 8; ++r) {
        const int m     = mt * 16 + half * 8 + r;
        const size_t ci = ((size_t)b * L + m) * F + f;
        float ig = hsig(acc[0][r]);
        float fg = hsig(acc[1][r]);
        float cg = fmaxf(acc[2][r], 0.f);
        float og = hsig(acc[3][r]);
        float cp = (t == 0) ? 0.f : cws[ci];
        float c  = fg * cp + ig * cg;
        float h  = og * fmaxf(c, 0.f);
        cws[ci]       = c;
        hnxt[m * F + f] = (_Float16)h;
        if constexpr (LAST) {
          if (t == T - 1) hlast[ci] = h;
        } else {
          hseq[(((size_t)b * T + t) * L + m) * F + f] = (_Float16)h;
        }
      }
    }
    __syncthreads();   // protects h ping-pong + sx restage
  }
}

// ---------------------------------------------------------------------------
// BatchNorm (inference) kernels
// ---------------------------------------------------------------------------
__global__ void bn_inplace_f16(_Float16* __restrict__ h, const float* __restrict__ g,
                               const float* __restrict__ be, const float* __restrict__ mu,
                               const float* __restrict__ var, int C, size_t total) {
  for (size_t i = (size_t)blockIdx.x * blockDim.x + threadIdx.x; i < total;
       i += (size_t)gridDim.x * blockDim.x) {
    int c   = (int)(i % C);
    float x = (float)h[i];
    h[i] = (_Float16)((x - mu[c]) * rsqrtf(var[c] + 1e-3f) * g[c] + be[c]);
  }
}

__global__ void bn_f32(const float* __restrict__ in, float* __restrict__ out,
                       const float* __restrict__ g, const float* __restrict__ be,
                       const float* __restrict__ mu, const float* __restrict__ var,
                       int C, size_t total) {
  for (size_t i = (size_t)blockIdx.x * blockDim.x + threadIdx.x; i < total;
       i += (size_t)gridDim.x * blockDim.x) {
    int c = (int)(i % C);
    out[i] = (in[i] - mu[c]) * rsqrtf(var[c] + 1e-3f) * g[c] + be[c];
  }
}

// ---------------------------------------------------------------------------
// Dense GEMM: out[M,NPAD] = relu?(A[M,K] @ W + bias). W pre-swizzled f16.
// One wave per 16x16 tile, 8 waves/block.
// ---------------------------------------------------------------------------
template <bool RELU>
__global__ __launch_bounds__(256) void gemm_rowmajor(
    const float* __restrict__ A, const _Float16* __restrict__ Wf,
    const float* __restrict__ bias, float* __restrict__ out, int M, int K,
    int NPAD, int NVALID) {
  const int tid = threadIdx.x, lane = tid & 31, wave = tid >> 5;
  const int half = lane >> 4, col = lane & 15;
  const int MT = M >> 4, NT = NPAD >> 4;
  const int tile = blockIdx.x * 8 + wave;
  if (tile >= MT * NT) return;
  const int mt = tile / NT, nt = tile % NT;
  const int n  = nt * 16 + col;

  const v16h* __restrict__ Wv = (const v16h*)Wf;
  float bv = (n < NVALID) ? bias[n] : 0.f;
  v8f acc;
#pragma unroll
  for (int r = 0; r < 8; ++r) acc[r] = bv;

  const float* ap = A + (size_t)(mt * 16 + col) * K;
  const int KC = K >> 5;
  for (int kc = 0; kc < KC; ++kc) {
    if (kc + 8 < KC)  // keep B tiles warm in L2/L0 (global_prefetch)
      __builtin_prefetch((const void*)(Wv + (size_t)((kc + 8) * NT + nt) * 32 + lane), 0, 1);
    const float* p0 = ap + kc * 32 + half * 8;
    float4 x0 = *(const float4*)(p0);
    float4 x1 = *(const float4*)(p0 + 4);
    float4 x2 = *(const float4*)(p0 + 16);
    float4 x3 = *(const float4*)(p0 + 20);
    float xs[16] = {x0.x, x0.y, x0.z, x0.w, x1.x, x1.y, x1.z, x1.w,
                    x2.x, x2.y, x2.z, x2.w, x3.x, x3.y, x3.z, x3.w};
    v16h af;
#pragma unroll
    for (int j = 0; j < 16; ++j) af[j] = (_Float16)xs[j];
    v16h bf = Wv[(size_t)(kc * NT + nt) * 32 + lane];
    acc = wmma16x16x32(af, bf, acc);
  }
#pragma unroll
  for (int r = 0; r < 8; ++r) {
    int m   = mt * 16 + half * 8 + r;
    float v = acc[r];
    if (RELU) v = fmaxf(v, 0.f);
    out[(size_t)m * NPAD + n] = v;
  }
}

__global__ void softmax_rows(const float* __restrict__ logits, float* __restrict__ out) {
  int r = threadIdx.x;
  if (r >= 32) return;
  float v[5];
  float mx = -1e30f;
#pragma unroll
  for (int j = 0; j < 5; ++j) { v[j] = logits[r * 16 + j]; mx = fmaxf(mx, v[j]); }
  float s = 0.f;
#pragma unroll
  for (int j = 0; j < 5; ++j) { v[j] = __expf(v[j] - mx); s += v[j]; }
  float inv = 1.f / s;
#pragma unroll
  for (int j = 0; j < 5; ++j) out[r * 5 + j] = v[j] * inv;
}

// ---------------------------------------------------------------------------
extern "C" void kernel_launch(void* const* d_in, const int* in_sizes, int n_in,
                              void* d_out, int out_size, void* d_ws, size_t ws_size,
                              hipStream_t stream) {
  const float* x   = (const float*)d_in[0];
  const float* Wx1 = (const float*)d_in[1];
  const float* Wh1 = (const float*)d_in[2];
  const float* b1  = (const float*)d_in[3];
  const float* Wx2 = (const float*)d_in[4];
  const float* Wh2 = (const float*)d_in[5];
  const float* b2  = (const float*)d_in[6];
  const float* Wx3 = (const float*)d_in[7];
  const float* Wh3 = (const float*)d_in[8];
  const float* b3  = (const float*)d_in[9];
  const float* g1 = (const float*)d_in[10], *be1 = (const float*)d_in[11];
  const float* m1 = (const float*)d_in[12], *v1 = (const float*)d_in[13];
  const float* g2 = (const float*)d_in[14], *be2 = (const float*)d_in[15];
  const float* m2 = (const float*)d_in[16], *v2 = (const float*)d_in[17];
  const float* g3 = (const float*)d_in[18], *be3 = (const float*)d_in[19];
  const float* m3 = (const float*)d_in[20], *v3 = (const float*)d_in[21];
  const float* D1 = (const float*)d_in[22], *db1 = (const float*)d_in[23];
  const float* D2 = (const float*)d_in[24], *db2 = (const float*)d_in[25];
  const float* D3 = (const float*)d_in[26], *db3 = (const float*)d_in[27];

  const int B = 32, T = 32, L = 128;

  // ---- workspace carve-up ----
  char* base = (char*)d_ws;
  size_t off = 0;
  auto carve = [&](size_t bytes) -> char* {
    char* p = base + off;
    off = (off + bytes + 255) & ~(size_t)255;
    return p;
  };
  const size_t w1e = (size_t)7 * 16 * 512;    // [KC][NT][512] f16 elems
  const size_t w2e = (size_t)18 * 32 * 512;
  const size_t w3e = (size_t)36 * 64 * 512;
  const size_t d1e = (size_t)1024 * 64 * 512;
  const size_t d2e = (size_t)32 * 32 * 512;
  const size_t d3e = (size_t)16 * 1 * 512;

  _Float16* w1f = (_Float16*)carve(w1e * 2);
  _Float16* w2f = (_Float16*)carve(w2e * 2);
  _Float16* w3f = (_Float16*)carve(w3e * 2);
  _Float16* d1f = (_Float16*)carve(d1e * 2);
  _Float16* d2f = (_Float16*)carve(d2e * 2);
  _Float16* d3f = (_Float16*)carve(d3e * 2);
  _Float16* h1  = (_Float16*)carve((size_t)B * T * L * 64 * 2);
  _Float16* h2  = (_Float16*)carve((size_t)B * T * L * 128 * 2);
  float*    h3  = (float*)carve((size_t)B * L * 256 * 4);
  float*    cbf = (float*)carve((size_t)B * L * 256 * 4);
  float*    din = (float*)carve((size_t)B * L * 256 * 4);
  float*    a1  = (float*)carve((size_t)B * 1024 * 4);
  float*    a2  = (float*)carve((size_t)B * 512 * 4);
  float*    lg  = (float*)carve((size_t)B * 16 * 4);
  (void)ws_size; (void)n_in; (void)in_sizes; (void)out_size;

  auto gsz = [](size_t total) -> int {
    size_t g = (total + 255) / 256;
    return (int)(g > 4096 ? 4096 : g);
  };

  // ---- weight swizzle + f16 convert (all L2-resident afterwards) ----
  prep_rnn_w<<<gsz(w1e), 256, 0, stream>>>(Wx1, Wh1, w1f, 1, 64, 32, w1e);
  prep_rnn_w<<<gsz(w2e), 256, 0, stream>>>(Wx2, Wh2, w2f, 64, 128, 192, w2e);
  prep_rnn_w<<<gsz(w3e), 256, 0, stream>>>(Wx3, Wh3, w3f, 128, 256, 384, w3e);
  prep_dense_w<<<gsz(d1e), 256, 0, stream>>>(D1, d1f, 32768, 1024, 1024, d1e);
  prep_dense_w<<<gsz(d2e), 256, 0, stream>>>(D2, d2f, 1024, 512, 512, d2e);
  prep_dense_w<<<gsz(d3e), 256, 0, stream>>>(D3, d3f, 512, 5, 16, d3e);

  // ---- ConvLSTM stack (one WG per batch element, recurrence inside) ----
  convlstm_kernel<1, 64, 32, false><<<B, 256, 0, stream>>>(
      x, nullptr, w1f, b1, cbf, h1, nullptr, B, T);
  bn_inplace_f16<<<4096, 256, 0, stream>>>(h1, g1, be1, m1, v1, 64,
                                           (size_t)B * T * L * 64);
  convlstm_kernel<64, 128, 192, false><<<B, 256, 0, stream>>>(
      nullptr, h1, w2f, b2, cbf, h2, nullptr, B, T);
  bn_inplace_f16<<<4096, 256, 0, stream>>>(h2, g2, be2, m2, v2, 128,
                                           (size_t)B * T * L * 128);
  convlstm_kernel<128, 256, 384, true><<<B, 256, 0, stream>>>(
      nullptr, h2, w3f, b3, cbf, nullptr, h3, B, T);
  bn_f32<<<4096, 256, 0, stream>>>(h3, din, g3, be3, m3, v3, 256,
                                   (size_t)B * L * 256);

  // ---- dense head ----
  gemm_rowmajor<true><<<16, 256, 0, stream>>>(din, d1f, db1, a1, 32, 32768, 1024, 1024);
  gemm_rowmajor<true><<<8, 256, 0, stream>>>(a1, d2f, db2, a2, 32, 1024, 512, 512);
  gemm_rowmajor<false><<<1, 256, 0, stream>>>(a2, d3f, db3, lg, 32, 512, 16, 5);
  softmax_rows<<<1, 32, 0, stream>>>(lg, (float*)d_out);
}
